// Attention_73701638800011
// MI455X (gfx1250) — matrix operand order repaired and checked by reading, not compiled, not run on hardware
//
#include <hip/hip_runtime.h>
#include <hip/hip_bf16.h>

#define HH 1024
#define LL 1024
#define BB 32

typedef __attribute__((ext_vector_type(16))) _Float16 v16h;
typedef __attribute__((ext_vector_type(8)))  float    v8f;

// ---------------- u1 = W^T w1, u2 = W^T w2 (split over o, partials) ----------------
__global__ void k_project_w(const float* __restrict__ W, const float* __restrict__ mlp_w,
                            float* __restrict__ upart) {
    int h  = blockIdx.x * 256 + threadIdx.x;   // 0..1023
    int oc = blockIdx.y;                        // 0..7, o-chunk of 128
    float a1 = 0.f, a2 = 0.f;
    int o0 = oc * 128;
    for (int o = o0; o < o0 + 128; ++o) {
        float w = W[(size_t)o * HH + h];
        a1 = fmaf(w, mlp_w[o],      a1);
        a2 = fmaf(w, mlp_w[HH + o], a2);
    }
    upart[(size_t)oc * 2048 + h]        = a1;
    upart[(size_t)oc * 2048 + 1024 + h] = a2;
}

__global__ void k_project_reduce(const float* __restrict__ upart, float* __restrict__ u12) {
    int i = blockIdx.x * 256 + threadIdx.x;    // 0..2047 (u1 then u2)
    float a = 0.f;
#pragma unroll
    for (int c = 0; c < 8; ++c) a += upart[(size_t)c * 2048 + i];
    u12[i] = a;
}

// ---------------- logits[b,l] = relu(q.u1 + k.u2 + bias), one wave per row ----------------
__global__ void k_logits(const float* __restrict__ q, const float* __restrict__ k,
                         const float* __restrict__ u1, const float* __restrict__ u2,
                         const float* __restrict__ bias, float* __restrict__ logits) {
    __shared__ __align__(16) float su1[HH];
    __shared__ __align__(16) float su2[HH];
    int tid = threadIdx.x;
    for (int i = tid; i < HH; i += 256) { su1[i] = u1[i]; su2[i] = u2[i]; }
    __syncthreads();

    int lane = tid & 31;
    int wave = tid >> 5;
    int row  = blockIdx.x * 8 + wave;          // 0..BB*LL-1

    const float4* q4  = (const float4*)(q + (size_t)row * HH);
    const float4* k4  = (const float4*)(k + (size_t)row * HH);
    const float4* u14 = (const float4*)&su1[0];
    const float4* u24 = (const float4*)&su2[0];

    float accq = 0.f, acck = 0.f;
#pragma unroll
    for (int i = 0; i < 8; ++i) {
        float4 qv = q4[lane + i * 32];
        float4 kv = k4[lane + i * 32];
        float4 a  = u14[lane + i * 32];
        float4 b  = u24[lane + i * 32];
        accq += qv.x * a.x + qv.y * a.y + qv.z * a.z + qv.w * a.w;
        acck += kv.x * b.x + kv.y * b.y + kv.z * b.z + kv.w * b.w;
    }
    float r = accq + acck;
#pragma unroll
    for (int off = 16; off > 0; off >>= 1) r += __shfl_xor(r, off, 32);
    if (lane == 0) logits[row] = fmaxf(r + bias[0], 0.f);
}

// ---------------- softmax over L, one block (1024 thr) per batch ----------------
__global__ void k_softmax(const float* __restrict__ logits, float* __restrict__ score) {
    __shared__ float red[32];
    int b = blockIdx.x, tid = threadIdx.x;
    int lane = tid & 31, wave = tid >> 5;
    float x = logits[(size_t)b * LL + tid];

    float m = x;
#pragma unroll
    for (int off = 16; off > 0; off >>= 1) m = fmaxf(m, __shfl_xor(m, off, 32));
    if (lane == 0) red[wave] = m;
    __syncthreads();
    if (tid < 32) {
        float t = red[tid];
#pragma unroll
        for (int off = 16; off > 0; off >>= 1) t = fmaxf(t, __shfl_xor(t, off, 32));
        red[tid] = t;
    }
    __syncthreads();
    m = red[0];
    __syncthreads();

    float e = expf(x - m);
    float s = e;
#pragma unroll
    for (int off = 16; off > 0; off >>= 1) s += __shfl_xor(s, off, 32);
    if (lane == 0) red[wave] = s;
    __syncthreads();
    if (tid < 32) {
        float t = red[tid];
#pragma unroll
        for (int off = 16; off > 0; off >>= 1) t += __shfl_xor(t, off, 32);
        red[tid] = t;
    }
    __syncthreads();
    score[(size_t)b * LL + tid] = e / red[0];
}

// ---------------- s_part[b,h] = sum_{l in chunk} score[b,l]*value[b,l,h] ----------------
__global__ void k_wsum_part(const float* __restrict__ value, const float* __restrict__ score,
                            float* __restrict__ spart) {
    int h  = blockIdx.x * 256 + threadIdx.x;
    int b  = blockIdx.y;
    int l0 = blockIdx.z * 256;
    const float* vb = value + (size_t)b * LL * HH;
    const float* sb = score + (size_t)b * LL;
    float acc = 0.f;
#pragma unroll 4
    for (int l = l0; l < l0 + 256; ++l)
        acc = fmaf(sb[l], vb[(size_t)l * HH + h], acc);
    spart[((size_t)blockIdx.z * BB + b) * HH + h] = acc;
}

__global__ void k_wsum_reduce(const float* __restrict__ spart, float* __restrict__ s) {
    int i = blockIdx.x * 256 + threadIdx.x;    // 0..BB*HH-1
    s[i] = spart[i] + spart[(size_t)BB * HH + i]
         + spart[(size_t)2 * BB * HH + i] + spart[(size_t)3 * BB * HH + i];
}

// ---------------- result = s @ W^T  (32x1024x1024) via v_wmma_f32_16x16x32_f16 ----------------
__global__ void __launch_bounds__(32)
k_out_gemm(const float* __restrict__ s, const float* __restrict__ W, float* __restrict__ out) {
    int lane = threadIdx.x;
    int lo = lane & 15, hi = lane >> 4;
    int n0 = blockIdx.x * 16;                  // output column tile (o)
    int m0 = blockIdx.y * 16;                  // output row tile (batch)

    const float* srow = s + (size_t)(m0 + lo) * HH;   // A row m = m0+lo
    const float* wrow = W + (size_t)(n0 + lo) * HH;   // B col n = n0+lo  (W[n,k] row-contig in k)

    v8f c = {0.f, 0.f, 0.f, 0.f, 0.f, 0.f, 0.f, 0.f};

    for (int k0 = 0; k0 < HH; k0 += 32) {
        // A fragment: lanes 0-15 K=k0+0..7 / k0+16..23 ; lanes 16-31 K=k0+8..15 / k0+24..31
        const float* a0p = srow + k0 + (hi << 3);
        const float* a1p = a0p + 16;
        // B fragment: lanes 0-15 K=k0+0..7 / k0+16..23 ; lanes 16-31 K=k0+8..15 / k0+24..31
        const float* bp  = wrow + k0 + (hi << 3);

        v16h af, bf;
        float4 x0 = *(const float4*)(a0p);
        float4 x1 = *(const float4*)(a0p + 4);
        float4 x2 = *(const float4*)(a1p);
        float4 x3 = *(const float4*)(a1p + 4);
        af[0]=(_Float16)x0.x; af[1]=(_Float16)x0.y; af[2]=(_Float16)x0.z; af[3]=(_Float16)x0.w;
        af[4]=(_Float16)x1.x; af[5]=(_Float16)x1.y; af[6]=(_Float16)x1.z; af[7]=(_Float16)x1.w;
        af[8]=(_Float16)x2.x; af[9]=(_Float16)x2.y; af[10]=(_Float16)x2.z; af[11]=(_Float16)x2.w;
        af[12]=(_Float16)x3.x; af[13]=(_Float16)x3.y; af[14]=(_Float16)x3.z; af[15]=(_Float16)x3.w;

        float4 y0 = *(const float4*)(bp);
        float4 y1 = *(const float4*)(bp + 4);
        float4 y2 = *(const float4*)(bp + 16);
        float4 y3 = *(const float4*)(bp + 20);
        bf[0]=(_Float16)y0.x; bf[1]=(_Float16)y0.y; bf[2]=(_Float16)y0.z; bf[3]=(_Float16)y0.w;
        bf[4]=(_Float16)y1.x; bf[5]=(_Float16)y1.y; bf[6]=(_Float16)y1.z; bf[7]=(_Float16)y1.w;
        bf[8]=(_Float16)y2.x; bf[9]=(_Float16)y2.y; bf[10]=(_Float16)y2.z; bf[11]=(_Float16)y2.w;
        bf[12]=(_Float16)y3.x; bf[13]=(_Float16)y3.y; bf[14]=(_Float16)y3.z; bf[15]=(_Float16)y3.w;

        c = __builtin_amdgcn_wmma_f32_16x16x32_f16(false, af, false, bf,
                                                   (short)0, c, false, false);
    }

#pragma unroll
    for (int r = 0; r < 8; ++r) {
        int m = m0 + (hi << 3) + r;            // C layout: VGPR r -> M = r + 8*hi, N = lo
        out[(size_t)m * HH + n0 + lo] = c[r];
    }
}

extern "C" void kernel_launch(void* const* d_in, const int* in_sizes, int n_in,
                              void* d_out, int out_size, void* d_ws, size_t ws_size,
                              hipStream_t stream) {
    const float* query = (const float*)d_in[0];
    const float* key   = (const float*)d_in[1];
    const float* value = (const float*)d_in[2];
    const float* W     = (const float*)d_in[3];
    const float* mlp_w = (const float*)d_in[4];
    const float* mlp_b = (const float*)d_in[5];

    float* out    = (float*)d_out;
    float* result = out;                       // 32*1024
    float* score  = out + BB * LL;             // 32*1024

    float* ws     = (float*)d_ws;
    float* u12    = ws;                        // u1 @0, u2 @1024     (2048)
    float* logits = ws + 2048;                 // 32768
    float* spart  = ws + 2048 + 32768;         // 4*32768
    float* s      = spart + 4 * BB * HH;       // 32768
    float* upart  = s + BB * HH;               // 8*2048

    k_project_w     <<<dim3(HH / 256, 8),      256,  0, stream>>>(W, mlp_w, upart);
    k_project_reduce<<<dim3(8),                256,  0, stream>>>(upart, u12);
    k_logits        <<<dim3(BB * LL / 8),      256,  0, stream>>>(query, key, u12, u12 + 1024,
                                                                  mlp_b, logits);
    k_softmax       <<<dim3(BB),               1024, 0, stream>>>(logits, score);
    k_wsum_part     <<<dim3(HH / 256, BB, 4),  256,  0, stream>>>(value, score, spart);
    k_wsum_reduce   <<<dim3(BB * HH / 256),    256,  0, stream>>>(spart, s);
    k_out_gemm      <<<dim3(HH / 16, BB / 16), 32,   0, stream>>>(s, W, result);
}
